// TorchDotProductAttention_12266426597903
// MI455X (gfx1250) — matrix operand order repaired and checked by reading, not compiled, not run on hardware
//
#include <hip/hip_runtime.h>
#include <hip/hip_bf16.h>

// Flash-attention forward for MI455X (gfx1250), wave32 + V_WMMA_F32_16X16X32_BF16.
// Shapes fixed by the reference: S=2048, B=2, H=16, D=128, fp32 in/out.
// Round 4: exp2-domain softmax (native v_exp_f32), unroll-by-2 with constant
// buffer pointers, rcp epilogue. Async-to-LDS K staging + DPP16 reductions kept.

#define S_LEN 2048
#define BATCH 2
#define HEADS 16
#define DHEAD 128
#define SB    (BATCH * HEADS * DHEAD)      // 4096 floats between consecutive seq rows
#define NORM  0.08838834764831845f         // 1/sqrt(128)
#define LOG2E 1.4426950408889634f
#define QSCALE (NORM * LOG2E)              // scores produced in log2 domain
#define NT    (S_LEN / 32)                 // 64 key tiles

#define KFP 132  // LDS pitch (f32)  for K tile rows   [32 x 128], async-filled
#define VP  40   // LDS pitch (bf16) for V^T tile rows [128 x 32]
#define PP  40   // LDS pitch (bf16) for per-wave P    [16 x 32]

typedef __attribute__((ext_vector_type(16))) __bf16 v16bf;
typedef __attribute__((ext_vector_type(8)))  __bf16 v8bf;
typedef __attribute__((ext_vector_type(8)))  float  v8f;
typedef int v4i __attribute__((vector_size(16)));   // matches async builtin param type

#define AS1 __attribute__((address_space(1)))
#define AS3 __attribute__((address_space(3)))

#if defined(__has_builtin)
#  if __has_builtin(__builtin_amdgcn_global_load_async_to_lds_b128)
#    define HAVE_ASYNC 1
#  endif
#endif
#ifndef HAVE_ASYNC
#  define HAVE_ASYNC 0
#endif

__device__ __forceinline__ void wait_async() {
#if defined(__has_builtin) && __has_builtin(__builtin_amdgcn_s_wait_asynccnt)
  __builtin_amdgcn_s_wait_asynccnt(0);
#else
  asm volatile("s_wait_asynccnt 0" ::: "memory");
#endif
}

__device__ __forceinline__ float fast_exp2(float x) {
#if defined(__has_builtin) && __has_builtin(__builtin_amdgcn_exp2f)
  return __builtin_amdgcn_exp2f(x);    // single v_exp_f32
#else
  return exp2f(x);
#endif
}
__device__ __forceinline__ float fast_rcp(float x) {
#if defined(__has_builtin) && __has_builtin(__builtin_amdgcn_rcpf)
  return __builtin_amdgcn_rcpf(x);     // single v_rcp_f32
#else
  return 1.0f / x;
#endif
}

// ---- DPP16 butterfly reduction over 16-lane half-wave (VALU only, no LDS) ----
// masks {1,2,7,15} are linearly independent -> full 16-lane reduction.
template <int CTRL>
__device__ __forceinline__ float dppf(float v) {
  return __builtin_bit_cast(
      float, __builtin_amdgcn_update_dpp(0, __builtin_bit_cast(int, v), CTRL,
                                         0xf, 0xf, true));
}
__device__ __forceinline__ float rmax16(float v) {
  v = fmaxf(v, dppf<0xB1>(v));   // quad_perm(1,0,3,2)  : xor 1
  v = fmaxf(v, dppf<0x4E>(v));   // quad_perm(2,3,0,1)  : xor 2
  v = fmaxf(v, dppf<0x141>(v));  // row_half_mirror     : xor 7
  v = fmaxf(v, dppf<0x140>(v));  // row_mirror          : xor 15
  return v;
}
__device__ __forceinline__ float rsum16(float v) {
  v += dppf<0xB1>(v);
  v += dppf<0x4E>(v);
  v += dppf<0x141>(v);
  v += dppf<0x140>(v);
  return v;
}

// Build bf16 fragment from two 16B bf16 LDS regions.
__device__ __forceinline__ v16bf ld_frag(const __bf16* p0, const __bf16* p1) {
  v8bf lo = *(const v8bf*)p0;
  v8bf hi = *(const v8bf*)p1;
  v16bf r;
#pragma unroll
  for (int i = 0; i < 8; ++i) { r[i] = lo[i]; r[i + 8] = hi[i]; }
  return r;
}

// 16 f32 -> v16bf (packs into v_cvt_pk_bf16_f32).
__device__ __forceinline__ v16bf cvt16(float4 a, float4 b, float4 c, float4 d) {
  v16bf r;
  r[0] = (__bf16)a.x;  r[1] = (__bf16)a.y;  r[2]  = (__bf16)a.z;  r[3]  = (__bf16)a.w;
  r[4] = (__bf16)b.x;  r[5] = (__bf16)b.y;  r[6]  = (__bf16)b.z;  r[7]  = (__bf16)b.w;
  r[8] = (__bf16)c.x;  r[9] = (__bf16)c.y;  r[10] = (__bf16)c.z;  r[11] = (__bf16)c.w;
  r[12] = (__bf16)d.x; r[13] = (__bf16)d.y; r[14] = (__bf16)d.z;  r[15] = (__bf16)d.w;
  return r;
}

// ---- staging helpers -------------------------------------------------------
// K tile: 32 keys x 128 d, f32 row-major, async global->LDS (ASYNCcnt engine).
__device__ __forceinline__ void kick_K_tile(const float* Kg, float* sKbuf,
                                            int kb, int tid) {
  const int key  = tid >> 2;
  const int dseg = (tid & 3) * 32;
  const float* kr = Kg + (size_t)(kb + key) * SB + dseg;
  float* lr = sKbuf + key * KFP + dseg;
#pragma unroll
  for (int c = 0; c < 8; ++c) {
#if HAVE_ASYNC
    __builtin_amdgcn_global_load_async_to_lds_b128(
        (AS1 v4i*)(kr + 4 * c),
        (AS3 v4i*)(lr + 4 * c), 0, 0);
#else
    *(float4*)(lr + 4 * c) = *(const float4*)(kr + 4 * c);
#endif
  }
}

// V tile: 32 keys x 128 d -> LDS TRANSPOSED [d][key] bf16 (feeds PV B-frags).
__device__ __forceinline__ void stage_V_tile(const float* Vg, __bf16* sVbuf,
                                             int kb, int tid) {
  const int key  = tid & 31;
  const int dseg = (tid >> 5) * 32;
  const float* vr = Vg + (size_t)(kb + key) * SB + dseg;
#pragma unroll
  for (int j = 0; j < 8; ++j) {
    float4 a = *(const float4*)(vr + 4 * j);
    sVbuf[(dseg + 4 * j + 0) * VP + key] = (__bf16)a.x;
    sVbuf[(dseg + 4 * j + 1) * VP + key] = (__bf16)a.y;
    sVbuf[(dseg + 4 * j + 2) * VP + key] = (__bf16)a.z;
    sVbuf[(dseg + 4 * j + 3) * VP + key] = (__bf16)a.w;
  }
}

__launch_bounds__(128)
__global__ void fa_fwd_gfx1250(const float* __restrict__ Q, const float* __restrict__ K,
                               const float* __restrict__ V, float* __restrict__ O) {
  __shared__ __align__(16) float  sKF[2 * 32 * KFP];     // double-buffered K (f32)
  __shared__ __align__(16) __bf16 sVT[2 * DHEAD * VP];   // double-buffered V^T (bf16)
  __shared__ __align__(16) __bf16 sP [4 * 16 * PP];      // per-wave P (bf16)

  const int tid  = threadIdx.x;
  const int lane = tid & 31;
  const int wv   = tid >> 5;
  const int l15  = lane & 15;
  const int hi   = lane >> 4;

  const int qtile = blockIdx.x;
  const int head  = blockIdx.y;
  const int batch = blockIdx.z;
  const int cbase = (batch * HEADS + head) * DHEAD;

  const float* Qg = Q + cbase;
  const float* Kg = K + cbase;
  const float* Vg = V + cbase;
  float*       Og = O + cbase;

  const int qbase = qtile * 64 + wv * 16;

  // ---- Q fragments: 16 rows x 128 d, scaled by log2(e)/sqrt(d), fp32 -> bf16 ----
  v16bf aq[4];
  {
    const float* qr = Qg + (size_t)(qbase + l15) * SB;
#pragma unroll
    for (int kk = 0; kk < 4; ++kk) {
      const int d0 = 32 * kk + 16 * hi;
      float4 f0 = *(const float4*)(qr + d0);
      float4 f1 = *(const float4*)(qr + d0 + 4);
      float4 f2 = *(const float4*)(qr + d0 + 8);
      float4 f3 = *(const float4*)(qr + d0 + 12);
      f0.x *= QSCALE; f0.y *= QSCALE; f0.z *= QSCALE; f0.w *= QSCALE;
      f1.x *= QSCALE; f1.y *= QSCALE; f1.z *= QSCALE; f1.w *= QSCALE;
      f2.x *= QSCALE; f2.y *= QSCALE; f2.z *= QSCALE; f2.w *= QSCALE;
      f3.x *= QSCALE; f3.y *= QSCALE; f3.z *= QSCALE; f3.w *= QSCALE;
      aq[kk] = cvt16(f0, f1, f2, f3);
    }
  }

  // ---- online softmax state + output accumulators (fp32, log2 domain) ----
  v8f acc[8];
  float m[8], lsum[8];
  {
    v8f z = {};
#pragma unroll
    for (int j = 0; j < 8; ++j) acc[j] = z;
#pragma unroll
    for (int r = 0; r < 8; ++r) { m[r] = -3.0e38f; lsum[r] = 0.0f; }
  }

  __bf16* sPw = &sP[wv * 16 * PP];

  // ---- prologue: stage tile 0 into buffer 0 ----
  kick_K_tile(Kg, sKF, 0, tid);
  stage_V_tile(Vg, sVT, 0, tid);
#if HAVE_ASYNC
  wait_async();
#endif
  __syncthreads();

  // One key-tile step; buffer pointers are compile-time-constant per call site.
  auto tile_step = [&](int it, const float* sKc, const __bf16* sVc,
                       float* sKn, __bf16* sVn) {
    // fire async copy of next K tile into the other buffer (overlaps compute)
    if (it + 1 < NT) kick_K_tile(Kg, sKn, (it + 1) * 32, tid);
    // prefetch V two tiles ahead into cache
    if (it + 2 < NT)
      __builtin_prefetch(Vg + (size_t)((it + 2) * 32 + (tid & 31)) * SB + (tid >> 5) * 32, 0, 1);

    // ---- S = Q K^T : two 16-key tiles, d=128 in 4 WMMA steps, pipelined feed ----
    v8f c0 = {}, c1 = {};
    {
      const float* kb0 = sKc + (l15)      * KFP + 16 * hi;  // key tile 0 row
      const float* kb1 = sKc + (16 + l15) * KFP + 16 * hi;  // key tile 1 row
      float4 x0 = *(const float4*)(kb0 + 0),  x1 = *(const float4*)(kb0 + 4);
      float4 x2 = *(const float4*)(kb0 + 8),  x3 = *(const float4*)(kb0 + 12);
      float4 y0 = *(const float4*)(kb1 + 0),  y1 = *(const float4*)(kb1 + 4);
      float4 y2 = *(const float4*)(kb1 + 8),  y3 = *(const float4*)(kb1 + 12);
#pragma unroll
      for (int kk = 0; kk < 4; ++kk) {
        float4 nx0 = x0, nx1 = x1, nx2 = x2, nx3 = x3;
        float4 ny0 = y0, ny1 = y1, ny2 = y2, ny3 = y3;
        if (kk < 3) {  // fetch next fragments while WMMA runs
          const float* p0 = kb0 + 32 * (kk + 1);
          const float* p1 = kb1 + 32 * (kk + 1);
          nx0 = *(const float4*)(p0 + 0); nx1 = *(const float4*)(p0 + 4);
          nx2 = *(const float4*)(p0 + 8); nx3 = *(const float4*)(p0 + 12);
          ny0 = *(const float4*)(p1 + 0); ny1 = *(const float4*)(p1 + 4);
          ny2 = *(const float4*)(p1 + 8); ny3 = *(const float4*)(p1 + 12);
        }
        v16bf b0 = cvt16(x0, x1, x2, x3);
        v16bf b1 = cvt16(y0, y1, y2, y3);
        c0 = __builtin_amdgcn_wmma_f32_16x16x32_bf16(false, aq[kk], false, b0, (short)0, c0, false, false);
        c1 = __builtin_amdgcn_wmma_f32_16x16x32_bf16(false, aq[kk], false, b1, (short)0, c1, false, false);
        x0 = nx0; x1 = nx1; x2 = nx2; x3 = nx3;
        y0 = ny0; y1 = ny1; y2 = ny2; y3 = ny3;
      }
    }

    // ---- online softmax (log2 domain, DPP16 reductions). Row M = r + 8*hi. ----
#pragma unroll
    for (int r = 0; r < 8; ++r) {
      float tm = rmax16(fmaxf(c0[r], c1[r]));
      float nm = fmaxf(m[r], tm);
      float f  = fast_exp2(m[r] - nm);
      float p0 = fast_exp2(c0[r] - nm);
      float p1 = fast_exp2(c1[r] - nm);
      float rs = rsum16(p0 + p1);
      lsum[r]  = lsum[r] * f + rs;
      m[r]     = nm;
#pragma unroll
      for (int j = 0; j < 8; ++j) acc[j][r] *= f;
      sPw[(r + 8 * hi) * PP + l15]      = (__bf16)p0;   // stage P for C->A transpose
      sPw[(r + 8 * hi) * PP + 16 + l15] = (__bf16)p1;
    }

    // ---- O += P V : pipelined B-fragment feed from transposed V^T LDS ----
    {
      const __bf16* pr = &sPw[l15 * PP];
      v16bf ap = ld_frag(pr + 16 * hi, pr + 16 * hi + 8);
      const __bf16* v0 = sVc + l15 * VP + 16 * hi;
      v16bf bv = ld_frag(v0, v0 + 8);
#pragma unroll
      for (int j = 0; j < 8; ++j) {
        v16bf nv = bv;
        if (j < 7) {
          const __bf16* vp = sVc + (16 * (j + 1) + l15) * VP + 16 * hi;
          nv = ld_frag(vp, vp + 8);
        }
        acc[j] = __builtin_amdgcn_wmma_f32_16x16x32_bf16(false, ap, false, bv, (short)0, acc[j], false, false);
        bv = nv;
      }
    }

    // ---- stage next V tile (manual transpose+cvt), close the iteration ----
    if (it + 1 < NT) stage_V_tile(Vg, sVn, (it + 1) * 32, tid);
#if HAVE_ASYNC
    wait_async();
#endif
    __syncthreads();
  };

#pragma unroll 1
  for (int it = 0; it < NT; it += 2) {   // NT is even; buffers alternate 0,1
    tile_step(it,     sKF,            sVT,             sKF + 32 * KFP, sVT + DHEAD * VP);
    tile_step(it + 1, sKF + 32 * KFP, sVT + DHEAD * VP, sKF,           sVT);
  }

  // ---- epilogue: O * (1/lsum) ----
#pragma unroll
  for (int r = 0; r < 8; ++r) {
    float inv = fast_rcp(lsum[r]);
    float* orow = Og + (size_t)(qbase + r + 8 * hi) * SB;
#pragma unroll
    for (int j = 0; j < 8; ++j) {
      orow[16 * j + l15] = acc[j][r] * inv;
    }
  }
}

extern "C" void kernel_launch(void* const* d_in, const int* in_sizes, int n_in,
                              void* d_out, int out_size, void* d_ws, size_t ws_size,
                              hipStream_t stream) {
  (void)in_sizes; (void)n_in; (void)out_size; (void)d_ws; (void)ws_size;
  const float* Q = (const float*)d_in[0];
  const float* K = (const float*)d_in[1];
  const float* V = (const float*)d_in[2];
  float* O = (float*)d_out;
  dim3 grid(S_LEN / 64, HEADS, BATCH);  // 32 x 16 x 2 = 1024 workgroups
  fa_fwd_gfx1250<<<grid, 128, 0, stream>>>(Q, K, V, O);
}
